// RNN_CRF_84146999263325
// MI455X (gfx1250) — matrix operand order repaired and checked by reading, not compiled, not run on hardware
//
#include <hip/hip_runtime.h>

typedef __bf16 bf16;
typedef __bf16 v16bf __attribute__((ext_vector_type(16)));
typedef __bf16 v8bf  __attribute__((ext_vector_type(8)));
typedef float  v8f   __attribute__((ext_vector_type(8)));
typedef unsigned int u32x4 __attribute__((ext_vector_type(4)));
typedef unsigned int u32x8 __attribute__((ext_vector_type(8)));

#define B_   64
#define T_   256
#define K_   128
#define E_   256
#define H_   512
#define NWG  16
#define NEGV (-10000.0f)

// ---- workspace layout (bytes) ----
#define WS_CTR    0u          // 1 uint (step barrier counter)
#define WS_HZERO  256u        // B*H bf16 zeros = 65536
#define WS_XBF    65792u      // B*T*E bf16 = 8 MB
#define WS_HS     8454400u    // T*B*H bf16 = 16 MB
#define WS_EMIS   25231616u   // T*B*K f32 = 8 MB

// ---- LSTM kernel LDS layout (exactly 320KB) ----
#define L_WSL   0u        // bf16[128][768] weight slice, 192KB
#define L_ASTX0 196608u   // bf16[64][256] x tile buf0, 32KB
#define L_ASTX1 229376u   // bf16[64][256] x tile buf1, 32KB
#define L_ASTH  262144u   // bf16[64][512] h tile, 64KB (z f32[64][128] aliases)

// Assemble a v16bf WMMA operand from two 16B-aligned 8-half chunks.
__device__ __forceinline__ v16bf ld_pair(const bf16* p0, const bf16* p1) {
  v8bf lo = *(const v8bf*)p0;
  v8bf hi = *(const v8bf*)p1;
  v16bf r;
#pragma unroll
  for (int i = 0; i < 8; ++i) { r[i] = lo[i]; r[i + 8] = hi[i]; }
  return r;
}

// Issue a TDM tensor_load_to_lds with a 2-group descriptor (<=2D tile).
// group0: count=1 | lds_addr | global_addr | type=2
// group1 words: w0 = data_size(2B)<<16 ; w1..w7 packed dims/strides.
__device__ __forceinline__ void tdm_issue(unsigned lds_off, unsigned long long g,
                                          unsigned w1, unsigned w2, unsigned w3,
                                          unsigned w4, unsigned w5, unsigned w6,
                                          unsigned w7) {
  u32x4 g0;
  g0[0] = 1u;                                         // count=1, user mode
  g0[1] = lds_off;                                    // LDS byte address
  g0[2] = (unsigned)g;                                // global_addr[31:0]
  g0[3] = (unsigned)((g >> 32) & 0x01FFFFFFu) | 0x80000000u;  // addr[56:32] | type=2
  u32x8 g1;
  g1[0] = 0x00010000u;  // workgroup_mask=0, data_size=1 (2 bytes)
  g1[1] = w1; g1[2] = w2; g1[3] = w3; g1[4] = w4; g1[5] = w5; g1[6] = w6; g1[7] = w7;
  asm volatile("tensor_load_to_lds %0, %1" : : "s"(g0), "s"(g1) : "memory");
}

// ---------------- kernel 0: init counter + zero h_init ----------------
__global__ __launch_bounds__(256) void init_kernel(unsigned int* ctr, bf16* hzero) {
  if (threadIdx.x == 0) *ctr = 0u;
  unsigned int* hz = (unsigned int*)hzero;
  for (int i = threadIdx.x; i < (B_ * H_) / 2; i += 256) hz[i] = 0u;
}

// ---------------- kernel 1: embedding -> X bf16 [b][t][e] ----------------
__global__ __launch_bounds__(256) void embed_kernel(
    const int* __restrict__ cx, const int* __restrict__ wx,
    const float* __restrict__ cemb, const float* __restrict__ wemb,
    bf16* __restrict__ Xbf) {
  int bt = blockIdx.x, e = threadIdx.x;
  float acc = wemb[(size_t)wx[bt] * E_ + e];
#pragma unroll
  for (int l = 0; l < 8; ++l)
    acc += 0.125f * cemb[(size_t)cx[bt * 8 + l] * E_ + e];
  Xbf[(size_t)bt * E_ + e] = (bf16)acc;
}

// ---------------- kernel 2: persistent LSTM (16 WGs, step-synced, TDM-staged) ----------------
__global__ __launch_bounds__(256) void lstm_kernel(
    const bf16* __restrict__ Xbf, const float* __restrict__ Wi,
    const float* __restrict__ Wh, const float* __restrict__ bias_p,
    const int* __restrict__ wx, const bf16* __restrict__ hzero,
    bf16* __restrict__ hs, unsigned int* __restrict__ ctr) {
  extern __shared__ char smem[];
  bf16*  Wsl  = (bf16*)(smem + L_WSL);    // [128][768], [n][k] transposed bf16
  bf16*  AstH = (bf16*)(smem + L_ASTH);   // [64][512]
  float* zbuf = (float*)(smem + L_ASTH);  // aliases AstH, [64][128] f32

  const int tid  = threadIdx.x;
  const int wg   = blockIdx.x;
  const int lane = tid & 31;
  const int w    = tid >> 5;   // wave id = N-tile id (8 waves)

  const unsigned asth_off  = (unsigned)(size_t)(smem + L_ASTH);
  const unsigned astx_off0 = (unsigned)(size_t)(smem + L_ASTX0);
  const unsigned astx_off1 = (unsigned)(size_t)(smem + L_ASTX1);

  // one-time: weight slice (bf16, [n][k] layout), bias regs, c regs = 0
  for (int idx = tid; idx < 128 * 768; idx += 256) {
    int n = idx / 768, k = idx - n * 768;
    int col = (n >> 5) * 512 + (wg << 5) + (n & 31);  // gate*512 + 32*wg + within
    float v = (k < 256) ? Wi[(size_t)k * 2048 + col] : Wh[(size_t)(k - 256) * 2048 + col];
    Wsl[n * 768 + k] = (bf16)v;
  }
  float bregs[4];
#pragma unroll
  for (int g = 0; g < 4; ++g) bregs[g] = bias_p[g * 512 + (wg << 5) + lane];
  float creg[8];
#pragma unroll
  for (int r = 0; r < 8; ++r) creg[r] = 0.f;

  // prefetch X tile for t=0 via TDM (2D tile: 64 rows x 256 bf16, row stride T*E)
  if (tid == 0)
    tdm_issue(astx_off0, (unsigned long long)(size_t)Xbf,
              0u, (1u | (16384u << 16)), (256u << 16), 64u, 65536u, 0u, 0u);
  __syncthreads();

  const int nb  = (w << 4) + (lane & 15);     // local N column
  const int kbB = (lane & 16) ? 8 : 0;        // B operand K sub-offset
  const int ra  = lane & 15;                  // A row within tile
  const int kbA = (lane & 16) ? 8 : 0;        // A operand K sub-offset

#pragma unroll 1
  for (int t = 0; t < T_; ++t) {
    const bf16* hsrc = (t == 0) ? hzero : (hs + (size_t)(t - 1) * B_ * H_);
    const bf16* AX = (bf16*)(smem + ((t & 1) ? L_ASTX1 : L_ASTX0));

    // TDM: h_{t-1} (contiguous 64KB, 1D) now; prefetch X[t+1] into the other buffer.
    if (tid == 0) {
      tdm_issue(asth_off, (unsigned long long)(size_t)hsrc,
                0u, (512u | (1u << 16)), 0x80000000u, 0u, 32768u, 0u, 0u);
      if (t + 1 < T_)
        tdm_issue((t & 1) ? astx_off0 : astx_off1,
                  (unsigned long long)(size_t)(Xbf + (size_t)(t + 1) * E_),
                  0u, (1u | (16384u << 16)), (256u << 16), 64u, 65536u, 0u, 0u);
      if (t + 1 < T_) __builtin_amdgcn_s_wait_tensorcnt(1);  // X[t+1] may stay in flight
      else            __builtin_amdgcn_s_wait_tensorcnt(0);
    }
    __syncthreads();

    // z = [x_t | h_{t-1}](64x768) x Wslice^T(768x128); wave w: one N tile, 4 M tiles
    v8f accs[4] = {v8f{}, v8f{}, v8f{}, v8f{}};
#pragma unroll 4
    for (int kx = 0; kx < 8; ++kx) {            // K = 0..255 from X tile
      const bf16* bp = Wsl + nb * 768 + (kx << 5) + kbB;
      v16bf bm = ld_pair(bp, bp + 16);
      v16bf am[4];
#pragma unroll
      for (int m = 0; m < 4; ++m) {
        const bf16* ap = AX + ((m << 4) + ra) * 256 + (kx << 5) + kbA;
        am[m] = ld_pair(ap, ap + 16);
      }
#pragma unroll
      for (int m = 0; m < 4; ++m)
        accs[m] = __builtin_amdgcn_wmma_f32_16x16x32_bf16(
            false, am[m], false, bm, (short)0, accs[m], false, false);
    }
#pragma unroll 4
    for (int kh = 0; kh < 16; ++kh) {           // K = 256..767 from H tile
      const bf16* bp = Wsl + nb * 768 + 256 + (kh << 5) + kbB;
      v16bf bm = ld_pair(bp, bp + 16);
      v16bf am[4];
#pragma unroll
      for (int m = 0; m < 4; ++m) {
        const bf16* ap = AstH + ((m << 4) + ra) * 512 + (kh << 5) + kbA;
        am[m] = ld_pair(ap, ap + 16);
      }
#pragma unroll
      for (int m = 0; m < 4; ++m)
        accs[m] = __builtin_amdgcn_wmma_f32_16x16x32_bf16(
            false, am[m], false, bm, (short)0, accs[m], false, false);
    }
    __syncthreads();   // all reads of AstH done -> safe to alias as zbuf

#pragma unroll
    for (int m = 0; m < 4; ++m)
#pragma unroll
      for (int v = 0; v < 8; ++v) {
        int zr = (m << 4) + v + ((lane & 16) ? 8 : 0);
        zbuf[zr * 128 + (w << 4) + (lane & 15)] = accs[m][v];
      }
    __syncthreads();

    // gates: thread owns hcl = lane, 8 batch rows; c kept in registers
#pragma unroll
    for (int r = 0; r < 8; ++r) {
      int b = (tid >> 5) + (r << 3);
      float zi = zbuf[b * 128 + lane]      + bregs[0];
      float zf = zbuf[b * 128 + 32 + lane] + bregs[1];
      float zg = zbuf[b * 128 + 64 + lane] + bregs[2];
      float zo = zbuf[b * 128 + 96 + lane] + bregs[3];
      float ig = 1.f / (1.f + __expf(-zi));
      float fg = 1.f / (1.f + __expf(-zf));
      float og = 1.f / (1.f + __expf(-zo));
      float gg = tanhf(zg);
      float cold = creg[r];
      float cnew = fg * cold + ig * gg;
      float hnew = og * tanhf(cnew);
      bool  mt   = wx[b * T_ + t] > 0;
      float hprev = (float)hsrc[b * H_ + (wg << 5) + lane];
      creg[r] = mt ? cnew : cold;
      hs[(size_t)t * B_ * H_ + b * H_ + (wg << 5) + lane] = (bf16)(mt ? hnew : hprev);
    }
    __syncthreads();
    __threadfence();   // release h slice to device scope
    if (tid == 0) {
      atomicAdd(ctr, 1u);
      unsigned int target = (unsigned int)(NWG * (t + 1));
      while (atomicAdd(ctr, 0u) < target) { __builtin_amdgcn_s_sleep(1); }
    }
    __syncthreads();
    __threadfence();   // acquire other WGs' h slices
  }
}

// ---------------- kernel 3: emissions = hs @ Wout + bout, masked ----------------
__global__ __launch_bounds__(256) void wout_kernel(
    const bf16* __restrict__ hs, const float* __restrict__ Wout,
    const float* __restrict__ bout, const int* __restrict__ wx,
    float* __restrict__ emis) {
  extern __shared__ char smem[];
  bf16*  Wsh = (bf16*)smem;              // 128*512 (WoutT [k][h])
  bf16*  Ash = (bf16*)(smem + 131072);   // 64*512
  float* bsh = (float*)(smem + 196608);  // 128
  const int tid = threadIdx.x, g = blockIdx.x;
  const int lane = tid & 31, w = tid >> 5;

  for (int idx = tid; idx < 128 * 512; idx += 256) {
    int n = idx >> 9, h = idx & 511;
    Wsh[idx] = (bf16)Wout[(size_t)h * 128 + n];
  }
  if (tid < 128) bsh[tid] = bout[tid];
  {
    const uint4* src = (const uint4*)(hs + (size_t)g * 64 * H_);
    uint4* dst = (uint4*)Ash;
    for (int c = tid; c < 64 * 64; c += 256) dst[c] = src[c];
  }
  __syncthreads();

  const int nb = (w << 4) + (lane & 15);
  const int kbB = (lane & 16) ? 8 : 0;
  const int ra = lane & 15;
  const int kbA = (lane & 16) ? 8 : 0;
  v8f accs[4] = {v8f{}, v8f{}, v8f{}, v8f{}};
#pragma unroll 4
  for (int ks = 0; ks < 16; ++ks) {
    const bf16* bp = Wsh + nb * 512 + (ks << 5) + kbB;
    v16bf bm = ld_pair(bp, bp + 16);
    v16bf am[4];
#pragma unroll
    for (int m = 0; m < 4; ++m) {
      const bf16* ap = Ash + ((m << 4) + ra) * 512 + (ks << 5) + kbA;
      am[m] = ld_pair(ap, ap + 16);
    }
#pragma unroll
    for (int m = 0; m < 4; ++m)
      accs[m] = __builtin_amdgcn_wmma_f32_16x16x32_bf16(
          false, am[m], false, bm, (short)0, accs[m], false, false);
  }
#pragma unroll
  for (int m = 0; m < 4; ++m)
#pragma unroll
    for (int v = 0; v < 8; ++v) {
      int r = (g << 6) + (m << 4) + v + ((lane & 16) ? 8 : 0);  // r = t*B + b
      int col = (w << 4) + (lane & 15);
      int tt = r >> 6, bb = r & 63;
      float mf = (wx[bb * T_ + tt] > 0) ? 1.f : 0.f;
      emis[(size_t)r * K_ + col] = (accs[m][v] + bsh[col]) * mf;
    }
}

// ---------------- kernel 4: CRF forward Z + gold score, out = Z - gold ----------------
__global__ __launch_bounds__(128) void crf_kernel(
    const float* __restrict__ emis, const float* __restrict__ trans,
    const int* __restrict__ wx, const int* __restrict__ y,
    float* __restrict__ out) {
  extern __shared__ char smem[];
  float* tr    = (float*)smem;             // 128*129 (padded vs bank conflicts)
  float* score = (float*)(smem + 66048);   // 128
  float* red   = (float*)(smem + 66560);   // 128
  int*   ired  = (int*)(smem + 67072);     // 128
  const int k = threadIdx.x, b = blockIdx.x;

  for (int idx = k; idx < K_ * K_; idx += 128) {
    int r = idx >> 7, c = idx & 127;
    tr[r * 129 + c] = trans[idx];
  }
  score[k] = (k == 1) ? 0.f : NEGV;  // SOS=1
  __syncthreads();

#pragma unroll 1
  for (int t = 0; t < T_; ++t) {
    const float* trk = tr + k * 129;
    float mx = -3.4e38f;
#pragma unroll 4
    for (int j = 0; j < 128; ++j) mx = fmaxf(mx, score[j] + trk[j]);
    float s = 0.f;
#pragma unroll 4
    for (int j = 0; j < 128; ++j) s += __expf(score[j] + trk[j] - mx);
    float em = emis[(size_t)(t * B_ + b) * K_ + k];
    float st = em + mx + __logf(s);
    float nv = (wx[b * T_ + t] > 0) ? st : score[k];
    __syncthreads();
    score[k] = nv;
    __syncthreads();
  }

  // Z = logsumexp_k(score[k] + trans[EOS=2][k])
  float v = score[k] + tr[2 * 129 + k];
  red[k] = v; __syncthreads();
  for (int s2 = 64; s2 > 0; s2 >>= 1) { if (k < s2) red[k] = fmaxf(red[k], red[k + s2]); __syncthreads(); }
  float mx = red[0]; __syncthreads();
  red[k] = __expf(v - mx); __syncthreads();
  for (int s2 = 64; s2 > 0; s2 >>= 1) { if (k < s2) red[k] += red[k + s2]; __syncthreads(); }
  float Z = mx + __logf(red[0]);

  // gold path score (parallel over t), lengths = sum(mask)
  float g = 0.f; int cnt = 0;
#pragma unroll 1
  for (int t = k; t < T_; t += 128) {
    if (wx[b * T_ + t] > 0) {
      int yp = y[b * 257 + t], yn = y[b * 257 + t + 1];
      g += emis[(size_t)(t * B_ + b) * K_ + yn] + tr[yn * 129 + yp];
      cnt += 1;
    }
  }
  __syncthreads();
  red[k] = g; ired[k] = cnt; __syncthreads();
  for (int s2 = 64; s2 > 0; s2 >>= 1) {
    if (k < s2) { red[k] += red[k + s2]; ired[k] += ired[k + s2]; }
    __syncthreads();
  }
  if (k == 0) {
    int len = ired[0];
    int last = y[b * 257 + len];
    out[b] = Z - (red[0] + tr[2 * 129 + last]);
  }
}

extern "C" void kernel_launch(void* const* d_in, const int* in_sizes, int n_in,
                              void* d_out, int out_size, void* d_ws, size_t ws_size,
                              hipStream_t stream) {
  const int*   cx    = (const int*)d_in[0];
  const int*   wx    = (const int*)d_in[1];
  const int*   y     = (const int*)d_in[2];
  const float* cemb  = (const float*)d_in[3];
  const float* wemb  = (const float*)d_in[4];
  const float* Wi    = (const float*)d_in[5];
  const float* Wh    = (const float*)d_in[6];
  const float* bias  = (const float*)d_in[7];
  const float* Wout  = (const float*)d_in[8];
  const float* bout  = (const float*)d_in[9];
  const float* trans = (const float*)d_in[10];
  float* out = (float*)d_out;

  char* ws = (char*)d_ws;
  unsigned int* ctr = (unsigned int*)(ws + WS_CTR);
  bf16*  hzero = (bf16*)(ws + WS_HZERO);
  bf16*  Xbf   = (bf16*)(ws + WS_XBF);
  bf16*  hs    = (bf16*)(ws + WS_HS);
  float* emis  = (float*)(ws + WS_EMIS);

  init_kernel<<<1, 256, 0, stream>>>(ctr, hzero);
  embed_kernel<<<B_ * T_, 256, 0, stream>>>(cx, wx, cemb, wemb, Xbf);
  lstm_kernel<<<NWG, 256, 327680, stream>>>(Xbf, Wi, Wh, bias, wx, hzero, hs, ctr);
  wout_kernel<<<(T_ * B_) / 64, 256, 197120, stream>>>(hs, Wout, bout, wx, emis);
  crf_kernel<<<B_, 128, 67584, stream>>>(emis, trans, wx, y, out);
}
